// LiquidS4_2319282340652
// MI455X (gfx1250) — matrix operand order repaired and checked by reading, not compiled, not run on hardware
//
#include <hip/hip_runtime.h>

// LiquidS4 pipeline for MI455X (gfx1250, wave32).
// - Big GEMMs + flash attention: v_wmma_f32_16x16x32_f16 (f16 in, f32 accum),
//   B tiles staged via async global->LDS DMA, double buffered.
// - Sequential liquid scan: full-f32 recurrence via v_wmma_f32_16x16x4_f32,
//   weights + h state resident in LDS (async DMA), 32-WG grid sync per step.

typedef __attribute__((ext_vector_type(16))) _Float16 v16h;
typedef __attribute__((ext_vector_type(8)))  float    v8f;
typedef __attribute__((ext_vector_type(2)))  float    v2f;

#define DEV static __device__ __forceinline__

union AFrag { v16h v; unsigned u[8]; _Float16 h[16]; };

// ---- CDNA5 async global->LDS copy (one 16B chunk per lane) -----------------
// LDS address operand = low 32 bits of the generic shared pointer
DEV unsigned lds_off(const void* p) {
  return (unsigned)(unsigned long long)p;
}
DEV void async_b128(unsigned ldst, const void* gsrc) {
  asm volatile("global_load_async_to_lds_b128 %0, %1, off"
               :: "v"(ldst), "v"((unsigned long long)gsrc)
               : "memory");
}
DEV void wait_async_le1() { asm volatile("s_wait_asynccnt 0x1" ::: "memory"); }
DEV void wait_async_0()   { asm volatile("s_wait_asynccnt 0x0" ::: "memory"); }

// Intra-wave LDS producer->consumer ordering (CDNA5 split counters).
DEV void lds_fence() { asm volatile("s_wait_dscnt 0" ::: "memory"); }

// ---- WMMA fragment loaders (wave32 layouts per CDNA5 ISA 7.12.2) -----------

// 16-bit A matrix 16x32: lane m = lane&15, half = lane>>4.
// VGPR i holds K pair k0 = ((i&4)?16:0) + 2*(i&3) + 8*half.
DEV v16h load_a_f16(const _Float16* p, int ldk) {
  const int lane = threadIdx.x & 31;
  const int m = lane & 15, half = lane >> 4;
  const _Float16* row = p + (size_t)m * ldk;
  AFrag f;
#pragma unroll
  for (int i = 0; i < 8; ++i) {
    const int k0 = ((i & 4) << 2) + ((i & 3) << 1) + (half << 3);
    f.u[i] = *(const unsigned*)(row + k0);
  }
  return f.v;
}

DEV v16h load_a_f32(const float* p, int ldk) {  // convert f32 -> f16 on the fly
  const int lane = threadIdx.x & 31;
  const int m = lane & 15, half = lane >> 4;
  const float* row = p + (size_t)m * ldk;
  AFrag f;
#pragma unroll
  for (int i = 0; i < 8; ++i) {
    const int k0 = ((i & 4) << 2) + ((i & 3) << 1) + (half << 3);
    f.h[2 * i]     = (_Float16)row[k0];
    f.h[2 * i + 1] = (_Float16)row[k0 + 1];
  }
  return f.v;
}

// 16-bit B matrix 32x16 loaded from B^T storage ([N][K] row-major, stride ldk):
// lane n = lane&15, VGPR j holds K pair k0 = ((j&4)?16:0) + 2*(j&3) + 8*half.
DEV v16h load_bt(const _Float16* bt, int ldk) {
  const int lane = threadIdx.x & 31;
  const int n = lane & 15, half = lane >> 4;
  const _Float16* row = bt + (size_t)n * ldk;
  AFrag f;
#pragma unroll
  for (int j = 0; j < 8; ++j) f.u[j] = *(const unsigned*)(row + ((j & 4) << 2) + ((j & 3) << 1) + (half << 3));
  return f.v;
}

// 32-bit 16x4 A (or 4x16 B from [n][k] storage): lane row = lane&15,
// VGPR0 = K(2*half), VGPR1 = K(2*half+1) -> one contiguous float2 per lane.
DEV v2f load_f2(const float* base, int stride, int kk) {
  const int lane = threadIdx.x & 31;
  const int r = lane & 15, half = lane >> 4;
  const float* p = base + (size_t)r * stride + kk + (half << 1);
  v2f v; v.x = p[0]; v.y = p[1];
  return v;
}

DEV v8f wmma_f16(v16h a, v16h b, v8f c) {
  return __builtin_amdgcn_wmma_f32_16x16x32_f16(false, a, false, b, (short)0, c,
                                                false, false);
}
DEV v8f wmma_f32(v2f a, v2f b, v8f c) {
  return __builtin_amdgcn_wmma_f32_16x16x4_f32(false, a, false, b, (short)0, c,
                                               false, false);
}

// ---- Generic GEMM: C[16x64] per wave, A[M,K] x BT[N,K] -> out --------------
// The 64x32 B tile is shared by all 8 waves of a block (same nj), staged into
// LDS with double-buffered async copies. MODE 0: f32 out; MODE 1: f16 out;
// MODE 2: qkv split (q,k -> [b,h,s,d], v -> [b,h,d,s]).
template <int A_F32, int MODE>
__global__ __launch_bounds__(256) void gemm16x64(
    const void* __restrict__ Ap, int lda, const _Float16* __restrict__ BT,
    const float* __restrict__ bias, float* __restrict__ outf,
    _Float16* __restrict__ outh, _Float16* __restrict__ outq,
    _Float16* __restrict__ outk, _Float16* __restrict__ outv, int M, int N,
    int K) {
  __shared__ _Float16 lB[2][64 * 32];   // double-buffered B tile (2 x 4KB)

  const int tid = threadIdx.x;
  const int wave = blockIdx.x * (blockDim.x >> 5) + (tid >> 5);
  const int mtiles = M >> 4;            // multiple of 8 -> nj uniform per block
  const int mi = wave % mtiles;
  const int nj = wave / mtiles;
  if (nj >= (N >> 6)) return;
  const int lane = tid & 31, half = lane >> 4;

  v8f c0 = {}, c1 = {}, c2 = {}, c3 = {};
  const _Float16* a16 = (const _Float16*)Ap + (size_t)mi * 16 * lda;
  const float* a32 = (const float*)Ap + (size_t)mi * 16 * lda;

  // Async staging: thread -> (row, 16B chunk) of the 64x32-half tile.
  const int brow = tid >> 2;                  // 0..63
  const int bpart = (tid & 3) << 3;           // halves: 0,8,16,24
  const _Float16* gB = BT + (size_t)(nj * 64 + brow) * K + bpart;
  const unsigned lb = lds_off(&lB[0][brow * 32 + bpart]);

  async_b128(lb, gB);                          // prefetch k=0 into buf0
  for (int k = 0; k < K; k += 32) {
    const int cur = (k >> 5) & 1;
    if (k + 32 < K) {
      async_b128(lb + (unsigned)(cur ^ 1) * 4096u, gB + k + 32);
      wait_async_le1();                        // oldest copy (cur) complete
    } else {
      wait_async_0();
    }
    __syncthreads();                           // tile visible to all waves

    v16h a;
    if constexpr (A_F32) a = load_a_f32(a32 + k, lda);
    else a = load_a_f16(a16 + k, lda);
    const _Float16* bb = &lB[cur][0];
    c0 = wmma_f16(a, load_bt(bb + 0 * 16 * 32, 32), c0);
    c1 = wmma_f16(a, load_bt(bb + 1 * 16 * 32, 32), c1);
    c2 = wmma_f16(a, load_bt(bb + 2 * 16 * 32, 32), c2);
    c3 = wmma_f16(a, load_bt(bb + 3 * 16 * 32, 32), c3);
    __syncthreads();                           // done reading before overwrite
  }

  v8f acc[4] = {c0, c1, c2, c3};
#pragma unroll
  for (int t = 0; t < 4; ++t) {
#pragma unroll
    for (int j = 0; j < 8; ++j) {
      const int m = mi * 16 + j + (half << 3);        // C layout: M = j (+8 upper)
      const int n = nj * 64 + t * 16 + (lane & 15);   // N = lane
      float v = acc[t][j];
      if (bias) v += bias[n];
      if constexpr (MODE == 0) {
        outf[(size_t)m * N + n] = v;
      } else if constexpr (MODE == 1) {
        outh[(size_t)m * N + n] = (_Float16)v;
      } else {
        const int b = m >> 10, s = m & 1023;
        if (n < 512) {
          const int hh = n >> 6, d = n & 63;
          outq[((((size_t)b * 8 + hh) << 10) + s) * 64 + d] = (_Float16)v;
        } else if (n < 1024) {
          const int e = n - 512, hh = e >> 6, d = e & 63;
          outk[((((size_t)b * 8 + hh) << 10) + s) * 64 + d] = (_Float16)v;
        } else {
          const int e = n - 1024, hh = e >> 6, d = e & 63;
          outv[((((size_t)b * 8 + hh) * 64 + d) << 10) + s] = (_Float16)v;
        }
      }
    }
  }
}

// ---- Liquid scan: full f32 via V_WMMA_F32_16X16X4_F32 ----------------------
// 32 WGs, one 16-column slice each. A^T/K^T slices (f32, [n][k], rows padded
// to 516 floats for conflict-free b64 fragment loads) live in LDS; h (16x512
// f32, batch padded 8->16) ping-pongs in global, staged to LDS by async DMA.
// Waves: {A,K} x {k-half}; 64-deep f32 WMMA chains, 2 interleaved accumulators.
#define LROW 516
__global__ __launch_bounds__(128) void scan_kernel(
    const float* __restrict__ uB, const float* __restrict__ AT32,
    const float* __restrict__ KT32, float* __restrict__ hb0,
    float* __restrict__ hb1, float* __restrict__ hs_out,
    _Float16* __restrict__ hs16, int* __restrict__ sync) {
  __shared__ float lA[16 * LROW];
  __shared__ float lK[16 * LROW];
  __shared__ float lh[16 * LROW];
  __shared__ float part[4][16][16];

  const int wg = blockIdx.x, tid = threadIdx.x;
  const int wid = tid >> 5, lane = tid & 31, half = lane >> 4;
  const int mat = wid >> 1, khalf = wid & 1;

  // Async DMA the WG's two 32KB f32 weight slices into (padded) LDS rows.
  const float* gA = AT32 + (size_t)wg * 16 * 512;
  const float* gK = KT32 + (size_t)wg * 16 * 512;
  for (int i = tid; i < 16 * 128; i += 128) {
    const int row = i >> 7, ch = (i & 127) << 2;    // 4 floats per 16B chunk
    async_b128(lds_off(&lA[row * LROW + ch]), gA + row * 512 + ch);
    async_b128(lds_off(&lK[row * LROW + ch]), gK + row * 512 + ch);
  }
  wait_async_0();
  __syncthreads();

  const float* W = mat ? lK : lA;
  const int kk0 = khalf << 8;                       // 0 or 256

  for (int t = 0; t < 1024; ++t) {
    const float* hin = (t & 1) ? hb1 : hb0;
    float* hout = (t & 1) ? hb0 : hb1;
    for (int i = tid; i < 16 * 128; i += 128) {     // async stage h -> LDS
      const int row = i >> 7, ch = (i & 127) << 2;
      async_b128(lds_off(&lh[row * LROW + ch]), hin + row * 512 + ch);
    }
    wait_async_0();
    __syncthreads();

    v8f c0 = {}, c1 = {};
    for (int kk = kk0; kk < kk0 + 256; kk += 8) {
      c0 = wmma_f32(load_f2(lh, LROW, kk),     load_f2(W, LROW, kk),     c0);
      c1 = wmma_f32(load_f2(lh, LROW, kk + 4), load_f2(W, LROW, kk + 4), c1);
    }
    const v8f c = c0 + c1;
#pragma unroll
    for (int j = 0; j < 8; ++j) part[wid][j + (half << 3)][lane & 15] = c[j];
    __syncthreads();

    if (wid == 0) {                                 // combine + recurrence
#pragma unroll
      for (int j = 0; j < 8; ++j) {
        const int m = j + (half << 3);
        const int nl = lane & 15;
        const int n = wg * 16 + nl;
        const float hA = part[0][m][nl] + part[1][m][nl];
        const float hK = part[2][m][nl] + part[3][m][nl];
        float ubt = 0.f, ubp = 0.f;
        if (m < 8) {
          const size_t r = ((size_t)m << 10) + t;
          ubt = uB[r * 512 + n];
          if (t > 0) ubp = uB[(r - 1) * 512 + n];
        }
        const float hn = hA + ubt + hK * ubp;
        hout[m * 512 + n] = hn;
        if (m < 8) {
          const size_t r = ((size_t)m << 10) + t;
          hs_out[r * 512 + n] = hn;
          hs16[r * 512 + n] = (_Float16)hn;
        }
      }
    }
    __threadfence();
    __syncthreads();
    if (tid == 0) {
      __hip_atomic_fetch_add(sync, 1, __ATOMIC_RELEASE, __HIP_MEMORY_SCOPE_AGENT);
      while (__hip_atomic_load(sync, __ATOMIC_ACQUIRE,
                               __HIP_MEMORY_SCOPE_AGENT) < 32 * (t + 1))
        __builtin_amdgcn_s_sleep(1);
    }
    __syncthreads();
  }
}

// ---- Flash attention: 1 wave per 16-row q tile, 32-key chunks --------------
__global__ __launch_bounds__(128) void attn_kernel(
    const _Float16* __restrict__ q, const _Float16* __restrict__ kmat,
    const _Float16* __restrict__ vt, _Float16* __restrict__ attno) {
  __shared__ float sbuf[4][16][36];       // padded: conflict-free row scans
  __shared__ _Float16 pbuf[4][16][32];
  __shared__ float fbuf[4][16];

  const int wid = threadIdx.x >> 5, lane = threadIdx.x & 31, half = lane >> 4;
  const int g = blockIdx.x * 4 + wid;
  const int bh = g >> 6, qt = g & 63;
  const int b = bh >> 3, h = bh & 7;

  const _Float16* qp = q + ((size_t)bh * 1024 + qt * 16) * 64;
  const v16h a0 = load_a_f16(qp, 64);        // q scale folded into softmax
  const v16h a1 = load_a_f16(qp + 32, 64);

  v8f o0 = {}, o1 = {}, o2 = {}, o3 = {};
  float mrow = -3.0e38f, lrow = 0.f;

  const _Float16* kbh = kmat + (size_t)bh * 1024 * 64;  // [s][d] == B^T for QK
  const _Float16* vbh = vt + (size_t)bh * 64 * 1024;    // [d][s] == B^T for PV

  for (int kc = 0; kc < 1024; kc += 32) {
    const _Float16* kp0 = kbh + (size_t)kc * 64;
    const _Float16* kp1 = kp0 + 16 * 64;
    v8f s0 = {}, s1 = {};
    s0 = wmma_f16(a0, load_bt(kp0, 64), s0);
    s0 = wmma_f16(a1, load_bt(kp0 + 32, 64), s0);
    s1 = wmma_f16(a0, load_bt(kp1, 64), s1);
    s1 = wmma_f16(a1, load_bt(kp1 + 32, 64), s1);
#pragma unroll
    for (int j = 0; j < 8; ++j) {
      sbuf[wid][j + (half << 3)][lane & 15] = s0[j];
      sbuf[wid][j + (half << 3)][16 + (lane & 15)] = s1[j];
    }
    lds_fence();
    if (lane < 16) {  // per-row online softmax stats, row r = lane
      const int r = lane;
      float cm = mrow;
#pragma unroll
      for (int ci = 0; ci < 32; ++ci)
        cm = fmaxf(cm, sbuf[wid][r][ci] * 0.125f);
      const float fr = __expf(mrow - cm);
      float csum = 0.f;
#pragma unroll
      for (int ci = 0; ci < 32; ++ci) {
        const float p = __expf(sbuf[wid][r][ci] * 0.125f - cm);
        pbuf[wid][r][ci] = (_Float16)p;
        csum += p;
      }
      lrow = lrow * fr + csum;
      mrow = cm;
      fbuf[wid][r] = fr;
    }
    lds_fence();
#pragma unroll
    for (int j = 0; j < 8; ++j) {  // rescale running output
      const float fj = fbuf[wid][j + (half << 3)];
      o0[j] *= fj; o1[j] *= fj; o2[j] *= fj; o3[j] *= fj;
    }
    const v16h pa = load_a_f16(&pbuf[wid][0][0], 32);
    o0 = wmma_f16(pa, load_bt(vbh + (size_t)0 * 1024 + kc, 1024), o0);
    o1 = wmma_f16(pa, load_bt(vbh + (size_t)16 * 1024 + kc, 1024), o1);
    o2 = wmma_f16(pa, load_bt(vbh + (size_t)32 * 1024 + kc, 1024), o2);
    o3 = wmma_f16(pa, load_bt(vbh + (size_t)48 * 1024 + kc, 1024), o3);
  }
  if (lane < 16) fbuf[wid][lane] = 1.0f / lrow;
  lds_fence();
  v8f oo[4] = {o0, o1, o2, o3};
#pragma unroll
  for (int t = 0; t < 4; ++t) {
#pragma unroll
    for (int j = 0; j < 8; ++j) {
      const float inv = fbuf[wid][j + (half << 3)];
      const int srow = (b << 10) + qt * 16 + j + (half << 3);
      const int e = h * 64 + t * 16 + (lane & 15);
      attno[(size_t)srow * 512 + e] = (_Float16)(oo[t][j] * inv);
    }
  }
}

// ---- Small prep kernels ----------------------------------------------------
__global__ void convC_kernel(const float* __restrict__ s,
                             _Float16* __restrict__ d, int n) {
  int i = blockIdx.x * 256 + threadIdx.x;
  if (i < n) d[i] = (_Float16)s[i];
}
__global__ void convT_kernel(const float* __restrict__ s,
                             _Float16* __restrict__ d, int R, int C) {
  int i = blockIdx.x * 256 + threadIdx.x;
  if (i < R * C) {
    int r = i / C, c = i % C;
    d[(size_t)c * R + r] = (_Float16)s[i];
  }
}
__global__ void convT32_kernel(const float* __restrict__ s,
                               float* __restrict__ d, int R, int C) {
  int i = blockIdx.x * 256 + threadIdx.x;
  if (i < R * C) {
    int r = i / C, c = i % C;
    d[(size_t)c * R + r] = s[i];
  }
}
__global__ void zero_kernel(float* __restrict__ p, int n) {
  int i = blockIdx.x * 256 + threadIdx.x;
  if (i < n) p[i] = 0.f;
}

// ---- Host orchestration ----------------------------------------------------
extern "C" void kernel_launch(void* const* d_in, const int* in_sizes, int n_in,
                              void* d_out, int out_size, void* d_ws,
                              size_t ws_size, hipStream_t stream) {
  (void)in_sizes; (void)n_in; (void)out_size; (void)ws_size;
  const float* inputs = (const float*)d_in[0];
  const float* A = (const float*)d_in[1];
  const float* B = (const float*)d_in[2];
  const float* C = (const float*)d_in[3];
  const float* Km = (const float*)d_in[4];
  const float* W_in = (const float*)d_in[5];
  const float* b_in = (const float*)d_in[6];
  const float* W_out = (const float*)d_in[7];
  const float* b_out = (const float*)d_in[8];

  char* ws = (char*)d_ws;
  size_t off = 0;
  auto alloc = [&](size_t bytes) -> void* {
    void* p = ws + off;
    off += (bytes + 255) & ~(size_t)255;
    return p;
  };
  float* uB = (float*)alloc(8192ull * 512 * 4);
  float* AT32 = (float*)alloc(512ull * 512 * 4);     // A^T f32 [n][k]
  float* KT32 = (float*)alloc(512ull * 512 * 4);     // K^T f32 [n][k]
  _Float16* BT = (_Float16*)alloc(512ull * 256 * 2);
  _Float16* CT = (_Float16*)alloc(256ull * 512 * 2);
  _Float16* WinH = (_Float16*)alloc(1536ull * 512 * 2);
  _Float16* WoutH = (_Float16*)alloc(512ull * 512 * 2);
  _Float16* hs16 = (_Float16*)alloc(8192ull * 512 * 2);
  _Float16* qb = (_Float16*)alloc(8192ull * 512 * 2);
  _Float16* kb = (_Float16*)alloc(8192ull * 512 * 2);
  _Float16* vtb = (_Float16*)alloc(8192ull * 512 * 2);
  _Float16* attno = (_Float16*)alloc(8192ull * 512 * 2);
  _Float16* yb = (_Float16*)alloc(8192ull * 512 * 2);
  float* hb0 = (float*)alloc(16ull * 512 * 4);   // contiguous: hb0|hb1|sync
  float* hb1 = (float*)alloc(16ull * 512 * 4);
  int* sync = (int*)alloc(256);

  float* out_outputs = (float*)d_out;                 // [8,1024,256] f32
  float* out_hs = out_outputs + 8192ull * 256;        // [8,1024,512] f32

  // Weight prep: f16 [N][K] for the f16-WMMA GEMMs; f32 [N][K] for the scan.
  convC_kernel<<<(1536 * 512 + 255) / 256, 256, 0, stream>>>(W_in, WinH, 1536 * 512);
  convC_kernel<<<(512 * 512 + 255) / 256, 256, 0, stream>>>(W_out, WoutH, 512 * 512);
  convT32_kernel<<<(512 * 512 + 255) / 256, 256, 0, stream>>>(A, AT32, 512, 512);
  convT32_kernel<<<(512 * 512 + 255) / 256, 256, 0, stream>>>(Km, KT32, 512, 512);
  convT_kernel<<<(256 * 512 + 255) / 256, 256, 0, stream>>>(B, BT, 256, 512);
  convT_kernel<<<(512 * 256 + 255) / 256, 256, 0, stream>>>(C, CT, 512, 256);
  zero_kernel<<<(16 * 512 * 2 + 64 + 255) / 256, 256, 0, stream>>>(
      hb0, 16 * 512 * 2 + 64);  // hb0, hb1, sync

  // uB = inputs @ B : [8192,256] x [256,512] -> f32
  gemm16x64<1, 0><<<512, 256, 0, stream>>>(inputs, 256, BT, nullptr, uB,
                                           nullptr, nullptr, nullptr, nullptr,
                                           8192, 512, 256);
  // sequential liquid scan, full f32 (writes hidden_states + f16 copy)
  scan_kernel<<<32, 128, 0, stream>>>(uB, AT32, KT32, hb0, hb1, out_hs, hs16,
                                      sync);
  // qkv = hs @ W_in^T + b_in, split into q/k [b,h,s,d] and v^T [b,h,d,s]
  gemm16x64<0, 2><<<1536, 256, 0, stream>>>(hs16, 512, WinH, b_in, nullptr,
                                            nullptr, qb, kb, vtb, 8192, 1536,
                                            512);
  attn_kernel<<<1024, 128, 0, stream>>>(qb, kb, vtb, attno);
  // y = attn_out @ W_out^T + b_out
  gemm16x64<0, 1><<<512, 256, 0, stream>>>(attno, 512, WoutH, b_out, nullptr,
                                           yb, nullptr, nullptr, nullptr, 8192,
                                           512, 512);
  // outputs = y @ C -> f32 d_out
  gemm16x64<0, 0><<<256, 256, 0, stream>>>(yb, 512, CT, nullptr, out_outputs,
                                           nullptr, nullptr, nullptr, nullptr,
                                           8192, 256, 512);
}
